// LSTM_2594160247282
// MI455X (gfx1250) — matrix operand order repaired and checked by reading, not compiled, not run on hardware
//
#include <hip/hip_runtime.h>
#include <hip/hip_bf16.h>

// ---------------------------------------------------------------------------
// LSTM on MI455X (gfx1250), wave32 + v_wmma_f32_16x16x32_bf16.
//
//  - Split each W (1536x1024) into W_h (rows 0..1023) and W_x (rows 1024..1535)
//    since xc = concat(h, x).
//  - Phase A (parallel): pre[t][g][b][j] = x[t] @ W_x[g] + b[g]  (one big GEMM)
//  - Phase B (sequential, 512 launches): per step, 4-gate GEMM
//    h_{t-1}[64x1024] @ W_h[1024 x 4096] accumulated on top of pre[t],
//    then fused sigmoid/tanh gate math -> C, h.
//  - Weights pre-packed bf16 in the exact WMMA-B lane layout (2 x b128/lane).
//  - K-loops are software-pipelined in registers: loads for kt+1 issue before
//    the WMMAs of kt, so waits become partial and load latency is hidden on
//    the sequential critical path.
// ---------------------------------------------------------------------------

typedef __attribute__((ext_vector_type(16))) __bf16 v16bf;
typedef __attribute__((ext_vector_type(8)))  float  v8f;
typedef __attribute__((ext_vector_type(4)))  unsigned int uint4v;

#define T_STEPS 512
#define BATCH   64
#define DIN     512
#define HID     1024

// packed weight sizes (elements)
static constexpr size_t WH_ELEMS  = 4ull * 64 * 32 * 512;   // gates x n-tiles x k-tiles x 512
static constexpr size_t WX_ELEMS  = 4ull * 64 * 16 * 512;   // k-tiles = DIN/32 = 16
static constexpr size_t XB_ELEMS  = (size_t)T_STEPS * BATCH * DIN;
static constexpr size_t HB_ELEMS  = 2ull * BATCH * HID;     // double buffer
static constexpr size_t C_ELEMS   = (size_t)BATCH * HID;
static constexpr size_t BIAS_ELEMS= 4ull * HID;
static constexpr size_t PRE_ELEMS = (size_t)T_STEPS * 4 * BATCH * HID;

// byte offsets into workspace
static constexpr size_t WH_OFF  = 0;
static constexpr size_t WX_OFF  = WH_OFF + WH_ELEMS * 2;
static constexpr size_t XB_OFF  = WX_OFF + WX_ELEMS * 2;
static constexpr size_t HB_OFF  = XB_OFF + XB_ELEMS * 2;
static constexpr size_t CW_OFF  = HB_OFF + HB_ELEMS * 2;
static constexpr size_t BI_OFF  = CW_OFF + C_ELEMS * 4;
static constexpr size_t PRE_OFF = BI_OFF + BIAS_ELEMS * 4;

union TileU {
    uint4v q[2];
    v16bf  v;
};

// Load a 16-bit WMMA operand from two (possibly non-contiguous) 16B chunks.
__device__ __forceinline__ v16bf load2x16(const __bf16* p0, const __bf16* p1) {
    TileU u;
    u.q[0] = *reinterpret_cast<const uint4v*>(p0);
    u.q[1] = *reinterpret_cast<const uint4v*>(p1);
    return u.v;
}

__device__ __forceinline__ v8f wmma_bf16(v16bf a, v16bf b, v8f c) {
    return __builtin_amdgcn_wmma_f32_16x16x32_bf16(
        /*neg_a=*/false, a, /*neg_b=*/false, b,
        /*c_mod=*/(short)0, c, /*reuse_a=*/false, /*reuse_b=*/false);
}

__device__ __forceinline__ float sigmoidf_fast(float x) {
    return 1.0f / (1.0f + __expf(-x));
}

// ---------------------------------------------------------------------------
// Pack W_h and W_x (f32, row-major (1536,1024)) into bf16 WMMA-B lane layout:
//   tile(kt,nt): element(lane, i) = W[k0 + (i<8 ? (lane>>4)*8 + i : 16 + (lane>>4)*8 + i-8)][n0 + (lane&15)]
// ---------------------------------------------------------------------------
__global__ void lstm_pack_w(const float* __restrict__ Wi, const float* __restrict__ Wf,
                            const float* __restrict__ Wo, const float* __restrict__ Wg,
                            __bf16* __restrict__ WhP, __bf16* __restrict__ WxP)
{
    long id = (long)blockIdx.x * blockDim.x + threadIdx.x;
    if (id < (long)WH_ELEMS) {
        long r = id;
        int i    = (int)(r & 15); r >>= 4;
        int lane = (int)(r & 31); r >>= 5;
        int kt   = (int)(r % 32); r /= 32;
        int nt   = (int)(r % 64); r /= 64;
        int g    = (int)r;
        int k = kt * 32 + (i < 8 ? (lane >> 4) * 8 + i : 16 + (lane >> 4) * 8 + (i - 8));     // h rows 0..1023
        int n = nt * 16 + (lane & 15);
        const float* W = (g == 0) ? Wi : (g == 1) ? Wf : (g == 2) ? Wo : Wg;
        WhP[id] = (__bf16)W[(size_t)k * HID + n];
    } else if (id < (long)(WH_ELEMS + WX_ELEMS)) {
        long r = id - (long)WH_ELEMS;
        long keep = r;
        int i    = (int)(r & 15); r >>= 4;
        int lane = (int)(r & 31); r >>= 5;
        int kt   = (int)(r % 16); r /= 16;
        int nt   = (int)(r % 64); r /= 64;
        int g    = (int)r;
        int k = kt * 32 + (i < 8 ? (lane >> 4) * 8 + i : 16 + (lane >> 4) * 8 + (i - 8));     // x rows -> +HID
        int n = nt * 16 + (lane & 15);
        const float* W = (g == 0) ? Wi : (g == 1) ? Wf : (g == 2) ? Wo : Wg;
        WxP[keep] = (__bf16)W[(size_t)(HID + k) * HID + n];
    }
}

// ---------------------------------------------------------------------------
// Prep: x -> bf16, h0 -> bf16 buffer 0, C0 -> f32 working C, biases -> [4][H].
// ---------------------------------------------------------------------------
__global__ void lstm_prep(const float* __restrict__ x, const float* __restrict__ h0,
                          const float* __restrict__ C0,
                          const float* __restrict__ bi, const float* __restrict__ bf,
                          const float* __restrict__ bo, const float* __restrict__ bg,
                          __bf16* __restrict__ xb, __bf16* __restrict__ h_init,
                          float* __restrict__ Cws, float* __restrict__ biasws)
{
    long id = (long)blockIdx.x * blockDim.x + threadIdx.x;
    if (id < (long)XB_ELEMS) { xb[id] = (__bf16)x[id]; return; }
    id -= (long)XB_ELEMS;
    if (id < (long)C_ELEMS) {
        h_init[id] = (__bf16)h0[id];
        Cws[id]    = C0[id];
        return;
    }
    id -= (long)C_ELEMS;
    if (id < (long)BIAS_ELEMS) {
        int g  = (int)(id >> 10);
        int jj = (int)(id & 1023);
        const float* b = (g == 0) ? bi : (g == 1) ? bf : (g == 2) ? bo : bg;
        biasws[id] = b[jj];
    }
}

// ---------------------------------------------------------------------------
// Phase A: pre[t][g][b][j] = x[t,b,:] @ Wx[g][:,j] + bias[g][j]
// GEMM: M = 32768, N = 4096, K = 512. One wave = 64x16 output strip.
// Software-pipelined K loop (prologue / 15 steady iterations / epilogue).
// ---------------------------------------------------------------------------
__global__ void lstm_pre_gemm(const __bf16* __restrict__ xb,     // [32768][512]
                              const __bf16* __restrict__ WxP,    // packed
                              const float*  __restrict__ biasws, // [4][1024]
                              float* __restrict__ pre)           // [T][4][B][H]
{
    int tid  = threadIdx.x;
    int lane = tid & 31;
    int wave = tid >> 5;
    long wid = (long)blockIdx.x * 8 + wave;     // 0 .. 131071
    int mblock = (int)(wid >> 8);               // 0..511
    int ntile  = (int)(wid & 255);              // 0..255
    int g  = ntile >> 6;
    int jt = ntile & 63;
    int colL = lane & 15;
    int hi   = lane >> 4;
    int j = jt * 16 + colL;
    float bias = biasws[g * HID + j];

    const __bf16* bbase = WxP + ((size_t)(g * 64 + jt) * 16) * 512 + lane * 16;
    const __bf16* abase = xb + (size_t)(mblock * 64 + colL) * DIN + hi * 8;

    v8f acc[4] = {};

    // prologue: tiles for kt = 0
    v16bf bt = load2x16(bbase, bbase + 8);
    v16bf at[4];
#pragma unroll
    for (int mt = 0; mt < 4; ++mt) {
        const __bf16* ap = abase + (size_t)mt * 16 * DIN;
        at[mt] = load2x16(ap, ap + 16);
    }

    for (int kt = 0; kt < 15; ++kt) {
        // issue loads for kt+1 before consuming kt
        const __bf16* bpn = bbase + (size_t)(kt + 1) * 512;
        v16bf btn = load2x16(bpn, bpn + 8);
        v16bf atn[4];
#pragma unroll
        for (int mt = 0; mt < 4; ++mt) {
            const __bf16* ap = abase + (size_t)mt * 16 * DIN + (kt + 1) * 32;
            atn[mt] = load2x16(ap, ap + 16);
        }
#pragma unroll
        for (int mt = 0; mt < 4; ++mt) acc[mt] = wmma_bf16(at[mt], bt, acc[mt]);
        bt = btn;
#pragma unroll
        for (int mt = 0; mt < 4; ++mt) at[mt] = atn[mt];
    }
    // epilogue: kt = 15
#pragma unroll
    for (int mt = 0; mt < 4; ++mt) acc[mt] = wmma_bf16(at[mt], bt, acc[mt]);

#pragma unroll
    for (int mt = 0; mt < 4; ++mt) {
#pragma unroll
        for (int r = 0; r < 8; ++r) {
            int row = mblock * 64 + mt * 16 + r + hi * 8;  // flat t*64 + b
            int t = row >> 6;
            int b = row & 63;
            pre[(((size_t)t * 4 + g) * BATCH + b) * HID + j] = acc[mt][r] + bias;
        }
    }
}

// ---------------------------------------------------------------------------
// Phase B: one launch per timestep. 64 WGs x 256 threads (8 wave32s).
// Wave w -> gate g = w>>1, m-half mh = w&1 (two 16x16 tiles, K = 1024).
// Software-pipelined K loop; prefetch weight stream 2 tiles ahead.
// ---------------------------------------------------------------------------
__global__ void lstm_step(const __bf16* __restrict__ hprev,   // [64][1024] bf16
                          __bf16* __restrict__ hnext,         // [64][1024] bf16
                          const __bf16* __restrict__ WhP,     // packed [4][64][32][512]
                          const float*  __restrict__ pre_t,   // [4][64][1024]
                          float* __restrict__ Cws,            // [64][1024]
                          float* __restrict__ hs_out_t,       // [64][1024]
                          float* __restrict__ C_out,          // d_out tail
                          int isLast)
{
    __shared__ float gbuf[4][BATCH][16];   // 16 KB

    int tid  = threadIdx.x;
    int lane = tid & 31;
    int wave = tid >> 5;
    int jb   = blockIdx.x;          // 0..63
    int g    = wave >> 1;
    int mh   = wave & 1;
    int colL = lane & 15;
    int hi   = lane >> 4;
    int j    = jb * 16 + colL;
    int m0   = mh * 32;

    // seed accumulators with precomputed x-part (+bias)
    v8f acc0, acc1;
    const float* pg = pre_t + (size_t)g * BATCH * HID;
#pragma unroll
    for (int r = 0; r < 8; ++r) {
        acc0[r] = pg[(size_t)(m0 + r + hi * 8) * HID + j];
        acc1[r] = pg[(size_t)(m0 + 16 + r + hi * 8) * HID + j];
    }

    const __bf16* bbase = WhP + ((size_t)(g * 64 + jb) * 32) * 512 + lane * 16;
    const __bf16* a0base = hprev + (size_t)(m0 + colL) * HID + hi * 8;
    const __bf16* a1base = a0base + (size_t)16 * HID;

    // prologue: tiles for kt = 0
    v16bf bt = load2x16(bbase, bbase + 8);
    v16bf a0 = load2x16(a0base, a0base + 16);
    v16bf a1 = load2x16(a1base, a1base + 16);

    for (int kt = 0; kt < 31; ++kt) {
        // prefetch weight stream 2 tiles ahead into near cache
        if (kt + 2 < 32) __builtin_prefetch(bbase + (size_t)(kt + 2) * 512, 0, 3);
        // issue loads for kt+1 before consuming kt
        const __bf16* bpn  = bbase + (size_t)(kt + 1) * 512;
        const __bf16* a0pn = a0base + (kt + 1) * 32;
        const __bf16* a1pn = a1base + (kt + 1) * 32;
        v16bf btn = load2x16(bpn, bpn + 8);
        v16bf a0n = load2x16(a0pn, a0pn + 16);
        v16bf a1n = load2x16(a1pn, a1pn + 16);

        acc0 = wmma_bf16(a0, bt, acc0);
        acc1 = wmma_bf16(a1, bt, acc1);

        bt = btn; a0 = a0n; a1 = a1n;
    }
    // epilogue: kt = 31
    acc0 = wmma_bf16(a0, bt, acc0);
    acc1 = wmma_bf16(a1, bt, acc1);

    // stage gate pre-activations in LDS
#pragma unroll
    for (int r = 0; r < 8; ++r) {
        gbuf[g][m0 + r + hi * 8][colL]      = acc0[r];
        gbuf[g][m0 + 16 + r + hi * 8][colL] = acc1[r];
    }
    __syncthreads();

    // fused gate math: i=sigmoid(g0), f=sigmoid(g1), o=sigmoid(g2), c~=tanh(g3)
#pragma unroll
    for (int e = tid; e < BATCH * 16; e += 256) {
        int b = e >> 4;
        int c = e & 15;
        float ig = sigmoidf_fast(gbuf[0][b][c]);
        float fg = sigmoidf_fast(gbuf[1][b][c]);
        float og = sigmoidf_fast(gbuf[2][b][c]);
        float gg = tanhf(gbuf[3][b][c]);
        size_t idx = (size_t)b * HID + jb * 16 + c;
        float Cn = fg * Cws[idx] + ig * gg;
        float hn = og * tanhf(Cn);
        Cws[idx]      = Cn;
        hnext[idx]    = (__bf16)hn;
        hs_out_t[idx] = hn;
        if (isLast) C_out[idx] = Cn;
    }
}

// ---------------------------------------------------------------------------
extern "C" void kernel_launch(void* const* d_in, const int* in_sizes, int n_in,
                              void* d_out, int out_size, void* d_ws, size_t ws_size,
                              hipStream_t stream) {
    const float* x      = (const float*)d_in[0];
    const float* h0     = (const float*)d_in[1];
    const float* C0     = (const float*)d_in[2];
    const float* W_in   = (const float*)d_in[3];
    const float* b_in   = (const float*)d_in[4];
    const float* W_over = (const float*)d_in[5];
    const float* b_over = (const float*)d_in[6];
    const float* W_out  = (const float*)d_in[7];
    const float* b_out  = (const float*)d_in[8];
    const float* W_int  = (const float*)d_in[9];
    const float* b_int  = (const float*)d_in[10];

    char* ws = (char*)d_ws;
    __bf16* WhP    = (__bf16*)(ws + WH_OFF);
    __bf16* WxP    = (__bf16*)(ws + WX_OFF);
    __bf16* xb     = (__bf16*)(ws + XB_OFF);
    __bf16* hb     = (__bf16*)(ws + HB_OFF);
    float*  Cws    = (float*)(ws + CW_OFF);
    float*  biasws = (float*)(ws + BI_OFF);
    float*  pre    = (float*)(ws + PRE_OFF);

    float* out  = (float*)d_out;
    float* hs   = out;                                    // [T][B][H]
    float* Cout = out + (size_t)T_STEPS * BATCH * HID;    // [B][H]

    // Pack weights (gate order: in, over(forget), out, intern)
    {
        long total = (long)(WH_ELEMS + WX_ELEMS);
        int grid = (int)((total + 255) / 256);
        lstm_pack_w<<<grid, 256, 0, stream>>>(W_in, W_over, W_out, W_int, WhP, WxP);
    }
    // Convert x / init h,C / stage biases
    {
        long total = (long)(XB_ELEMS + C_ELEMS + BIAS_ELEMS);
        int grid = (int)((total + 255) / 256);
        lstm_prep<<<grid, 256, 0, stream>>>(x, h0, C0, b_in, b_over, b_out, b_int,
                                            xb, hb /*buffer 0*/, Cws, biasws);
    }
    // Phase A: parallel x-part GEMM for all timesteps (131072 waves / 8 = 16384 WGs)
    lstm_pre_gemm<<<16384, 256, 0, stream>>>(xb, WxP, biasws, pre);

    // Phase B: sequential steps (kernel boundary = global sync on h)
    for (int t = 0; t < T_STEPS; ++t) {
        const __bf16* hin = hb + (size_t)(t & 1) * BATCH * HID;
        __bf16* hout      = hb + (size_t)((t + 1) & 1) * BATCH * HID;
        lstm_step<<<64, 256, 0, stream>>>(
            hin, hout, WhP,
            pre + (size_t)t * 4 * BATCH * HID,
            Cws,
            hs + (size_t)t * BATCH * HID,
            Cout,
            (t == T_STEPS - 1) ? 1 : 0);
    }
}
